// OrganicMetal_GNN_30150670418426
// MI455X (gfx1250) — compile-verified, matrix operand order checked by reading, not run on hardware
//
#include <hip/hip_runtime.h>
#include <hip/hip_bf16.h>

// ---------------------------------------------------------------------------
// MI455X (gfx1250) D-MPNN bond message passing.
// - bf16 edge-state storage (halves HBM traffic vs f32; 23.3 TB/s roofline)
// - v_wmma_f32_16x16x32_bf16 for all three GEMMs (f32 accumulate)
// - f32 hardware atomics for segment_sum (agg = 25.6MB, resident in 192MB L2)
// - per-block weight tables copied to LDS via gfx1250 async-to-LDS DMA
//   (global_load_async_to_lds_b128 + s_wait_asynccnt), WMMA B from ds_load
// ---------------------------------------------------------------------------

typedef __attribute__((ext_vector_type(16))) __bf16 v16bf;
typedef __attribute__((ext_vector_type(8)))  float  v8f;

#define GNN_N_NODES 50000
#define GNN_N_EDGES 600000
#define GNN_HID 128

union FragU { v16bf v; uint4 q[2]; };

// gfx1250 async copy: LDS[lds_ptr] = MEM[gbase + goff], 16 bytes per lane.
// GVS addressing: SGPR-pair base + per-lane 32-bit VGPR offset. Tracked by
// ASYNCcnt. LDS address = low 32 bits of the flat shared-memory pointer
// (ISA 10.2: LDS aperture maps addr[31:0] to the LDS offset).
__device__ __forceinline__ void async_gl_to_lds_b128(void* lds_ptr,
                                                     const void* gbase,
                                                     unsigned goff) {
  unsigned l = (unsigned)(unsigned long long)lds_ptr;
  asm volatile("global_load_async_to_lds_b128 %0, %1, %2"
               :: "v"(l), "v"(goff), "s"(gbase)
               : "memory");
}
__device__ __forceinline__ void wait_async0() {
  asm volatile("s_wait_asynccnt 0" ::: "memory");
}

// ---------------------------------------------------------------------------
// Pack weights into per-fragment, per-lane bf16 layout matching the CDNA5
// 16-bit B-matrix (B^T-in-A-layout) convention: fragment (kt,nt), lane l holds
// column n = nt*16 + (l&15), K chunk kb = (l>>4)*8:
//   elems 0..7  -> K = kt*32 + kb + i
//   elems 8..15 -> K = kt*32 + 16 + kb + (i-8)
// Wi: K padded 144->160 (5 ktiles), Wh: 4 ktiles, Wo: 8 ktiles. NT=8 each.
// ---------------------------------------------------------------------------
__global__ void k_pack_weights(const float* __restrict__ Wi,
                               const float* __restrict__ Wh,
                               const float* __restrict__ Wo,
                               __bf16* wiF, __bf16* whF, __bf16* woF) {
  int t = blockIdx.x * blockDim.x + threadIdx.x;
  if (t >= 136 * 32) return;
  int fi = t >> 5, lane = t & 31;
  const float* W; __bf16* F; int local, kmax;
  if (fi < 40)      { W = Wi; F = wiF; local = fi;      kmax = 144; }
  else if (fi < 72) { W = Wh; F = whF; local = fi - 40; kmax = 128; }
  else              { W = Wo; F = woF; local = fi - 72; kmax = 256; }
  int kt = local >> 3, nt = local & 7;
  int n  = nt * 16 + (lane & 15);
  int kb = (lane >> 4) * 8;
  __bf16* out = F + (size_t)(local * 32 + lane) * 16;
  for (int i = 0; i < 16; ++i) {
    int k = kt * 32 + ((i < 8) ? (kb + i) : (16 + kb + (i - 8)));
    float v = (k < kmax) ? W[(size_t)k * GNN_HID + n] : 0.0f;
    out[i] = (__bf16)v;
  }
}

// ---------------------------------------------------------------------------
// H0 = relu([x[src]; edge_attr] @ Wi + bi), K = 144 padded to 160.
// One 16-edge tile per wave; 8 waves per block. Wi fragments async-DMA'd to
// LDS once per block.
// ---------------------------------------------------------------------------
__global__ __launch_bounds__(256)
void k_edge_h0(const float* __restrict__ x, const float* __restrict__ eattr,
               const int* __restrict__ src, const float* __restrict__ bi,
               const __bf16* __restrict__ wiF,
               __bf16* __restrict__ H0, __bf16* __restrict__ H, int ntiles) {
  __shared__ __bf16 stage[8][16][160];      // 40960 B
  __shared__ __bf16 wlds[40 * 32 * 16];     // 40960 B
  int lane = threadIdx.x & 31, w = threadIdx.x >> 5;
  int tile = blockIdx.x * 8 + w;

  // async copy of packed Wi fragments into LDS (1280 x 16B chunks)
  for (int it = 0; it < 5; ++it) {
    int c = it * 256 + threadIdx.x;
    async_gl_to_lds_b128(&wlds[c * 8], wiF, (unsigned)(c * 16));
  }

  bool active = (tile < ntiles);
  if (active) {
    int e0 = tile * 16;
    // stage 16 rows x 160 bf16, 4-wide chunks (region boundaries 128/144 are
    // multiples of 4 so each chunk lives in one source)
    for (int it = 0; it < 20; ++it) {
      int q = it * 32 + lane;
      int r = q / 40;
      int c4 = (q - r * 40) * 4;
      int e = e0 + r;
      float4 v;
      if (c4 < 128) {
        int s = src[e];
        v = *(const float4*)(x + (size_t)s * 128 + c4);
      } else if (c4 < 144) {
        v = *(const float4*)(eattr + (size_t)e * 16 + (c4 - 128));
      } else {
        v = make_float4(0.f, 0.f, 0.f, 0.f);
      }
      __bf16* d = &stage[w][r][c4];
      d[0] = (__bf16)v.x; d[1] = (__bf16)v.y;
      d[2] = (__bf16)v.z; d[3] = (__bf16)v.w;
    }
  }
  wait_async0();
  __syncthreads();
  if (!active) return;

  v8f acc[8];
  for (int i = 0; i < 8; ++i) acc[i] = (v8f){0,0,0,0,0,0,0,0};
  int m = lane & 15, kb = (lane >> 4) * 8;
  for (int kt = 0; kt < 5; ++kt) {
    FragU a;
    a.q[0] = *(const uint4*)&stage[w][m][kt * 32 + kb];
    a.q[1] = *(const uint4*)&stage[w][m][kt * 32 + 16 + kb];
    for (int nt = 0; nt < 8; ++nt) {
      FragU b;
      const uint4* bp = (const uint4*)&wlds[(size_t)((kt * 8 + nt) * 32 + lane) * 16];
      b.q[0] = bp[0]; b.q[1] = bp[1];
      acc[nt] = __builtin_amdgcn_wmma_f32_16x16x32_bf16(
          false, a.v, false, b.v, (short)0, acc[nt], false, false);
    }
  }
  int e0 = tile * 16;
  for (int nt = 0; nt < 8; ++nt) {
    int n = nt * 16 + (lane & 15);
    float bias = bi[n];
    for (int r = 0; r < 8; ++r) {
      int M = r + (lane >> 4) * 8;
      float v = acc[nt][r] + bias;
      v = v > 0.f ? v : 0.f;
      __bf16 hv = (__bf16)v;
      size_t off = (size_t)(e0 + M) * GNN_HID + n;
      H0[off] = hv;
      H[off]  = hv;
    }
  }
}

// ---------------------------------------------------------------------------
// agg zeroing (float4 per thread)
// ---------------------------------------------------------------------------
__global__ void k_zero(float* __restrict__ agg, int nquads) {
  int t = blockIdx.x * blockDim.x + threadIdx.x;
  if (t < nquads) *(float4*)(agg + (size_t)t * 4) = make_float4(0.f, 0.f, 0.f, 0.f);
}

// ---------------------------------------------------------------------------
// segment_sum: agg[dst[e]] += H[e]   (f32 hardware atomics; agg is L2-resident)
// ---------------------------------------------------------------------------
__global__ void k_scatter(const __bf16* __restrict__ H, const int* __restrict__ dst,
                          float* __restrict__ agg, int total) {
  int t = blockIdx.x * blockDim.x + threadIdx.x;
  if (t >= total) return;
  int e = t >> 5, c4 = (t & 31) * 4;
  int d = dst[e];
  const __bf16* hp = H + (size_t)e * GNN_HID + c4;
  float* ap = agg + (size_t)d * GNN_HID + c4;
  unsafeAtomicAdd(ap + 0, (float)hp[0]);
  unsafeAtomicAdd(ap + 1, (float)hp[1]);
  unsafeAtomicAdd(ap + 2, (float)hp[2]);
  unsafeAtomicAdd(ap + 3, (float)hp[3]);
}

// ---------------------------------------------------------------------------
// H_out = relu(H0 + (agg[src] - H_in[rev]) @ Wh + bh), K = 128.
// ---------------------------------------------------------------------------
__global__ __launch_bounds__(256)
void k_edge_update(const float* __restrict__ agg, const __bf16* __restrict__ Hin,
                   const __bf16* __restrict__ H0,
                   const int* __restrict__ src, const int* __restrict__ rev,
                   const float* __restrict__ bh, const __bf16* __restrict__ whF,
                   __bf16* __restrict__ Hout, int ntiles) {
  __shared__ __bf16 stage[8][16][128];      // 32768 B
  __shared__ __bf16 wlds[32 * 32 * 16];     // 32768 B
  int lane = threadIdx.x & 31, w = threadIdx.x >> 5;
  int tile = blockIdx.x * 8 + w;

  // async copy of packed Wh fragments into LDS (1024 x 16B chunks)
  for (int it = 0; it < 4; ++it) {
    int c = it * 256 + threadIdx.x;
    async_gl_to_lds_b128(&wlds[c * 8], whF, (unsigned)(c * 16));
  }

  bool active = (tile < ntiles);
  if (active) {
    int e0 = tile * 16;
    for (int it = 0; it < 16; ++it) {
      int q = it * 32 + lane;
      int r = q >> 5;
      int c4 = (q & 31) * 4;
      int e = e0 + r;
      int s  = src[e];
      int re = rev[e];
      float4 av = *(const float4*)(agg + (size_t)s * GNN_HID + c4);
      const __bf16* hp = Hin + (size_t)re * GNN_HID + c4;
      __bf16* d = &stage[w][r][c4];
      d[0] = (__bf16)(av.x - (float)hp[0]);
      d[1] = (__bf16)(av.y - (float)hp[1]);
      d[2] = (__bf16)(av.z - (float)hp[2]);
      d[3] = (__bf16)(av.w - (float)hp[3]);
    }
  }
  wait_async0();
  __syncthreads();
  if (!active) return;

  v8f acc[8];
  for (int i = 0; i < 8; ++i) acc[i] = (v8f){0,0,0,0,0,0,0,0};
  int m = lane & 15, kb = (lane >> 4) * 8;
  for (int kt = 0; kt < 4; ++kt) {
    FragU a;
    a.q[0] = *(const uint4*)&stage[w][m][kt * 32 + kb];
    a.q[1] = *(const uint4*)&stage[w][m][kt * 32 + 16 + kb];
    for (int nt = 0; nt < 8; ++nt) {
      FragU b;
      const uint4* bp = (const uint4*)&wlds[(size_t)((kt * 8 + nt) * 32 + lane) * 16];
      b.q[0] = bp[0]; b.q[1] = bp[1];
      acc[nt] = __builtin_amdgcn_wmma_f32_16x16x32_bf16(
          false, a.v, false, b.v, (short)0, acc[nt], false, false);
    }
  }
  int e0 = tile * 16;
  for (int nt = 0; nt < 8; ++nt) {
    int n = nt * 16 + (lane & 15);
    float bias = bh[n];
    for (int r = 0; r < 8; ++r) {
      int M = r + (lane >> 4) * 8;
      size_t off = (size_t)(e0 + M) * GNN_HID + n;
      float v = (float)H0[off] + acc[nt][r] + bias;
      v = v > 0.f ? v : 0.f;
      Hout[off] = (__bf16)v;
    }
  }
}

// ---------------------------------------------------------------------------
// Nodes with zero incoming message fall back to x.
// ---------------------------------------------------------------------------
__global__ void k_fixup(const float* __restrict__ x, float* __restrict__ agg, int n_nodes) {
  int n = blockIdx.x * blockDim.x + threadIdx.x;
  if (n >= n_nodes) return;
  float s = 0.f;
  for (int i = 0; i < GNN_HID; ++i) s += agg[(size_t)n * GNN_HID + i];
  if (s == 0.f)
    for (int i = 0; i < GNN_HID; ++i) agg[(size_t)n * GNN_HID + i] = x[(size_t)n * 128 + i];
}

// ---------------------------------------------------------------------------
// out = relu([x; M] @ Wo + bo), K = 256. 2 waves/block (80 KB LDS with the
// async-staged Wo fragment table).
// ---------------------------------------------------------------------------
__global__ __launch_bounds__(64)
void k_node_out(const float* __restrict__ x, const float* __restrict__ agg,
                const float* __restrict__ bo, const __bf16* __restrict__ woF,
                float* __restrict__ out, int ntiles) {
  __shared__ __bf16 stage[2][16][256];      // 16384 B
  __shared__ __bf16 wlds[64 * 32 * 16];     // 65536 B
  int lane = threadIdx.x & 31, w = threadIdx.x >> 5;
  int tile = blockIdx.x * 2 + w;

  // async copy of packed Wo fragments into LDS (2048 x 16B chunks)
  for (int it = 0; it < 32; ++it) {
    int c = it * 64 + threadIdx.x;
    async_gl_to_lds_b128(&wlds[c * 8], woF, (unsigned)(c * 16));
  }

  bool active = (tile < ntiles);
  if (active) {
    int n0 = tile * 16;
    for (int it = 0; it < 32; ++it) {
      int q = it * 32 + lane;
      int r = q >> 6;
      int c4 = (q & 63) * 4;
      int node = n0 + r;
      float4 v = (c4 < 128)
          ? *(const float4*)(x   + (size_t)node * 128 + c4)
          : *(const float4*)(agg + (size_t)node * GNN_HID + (c4 - 128));
      __bf16* d = &stage[w][r][c4];
      d[0] = (__bf16)v.x; d[1] = (__bf16)v.y;
      d[2] = (__bf16)v.z; d[3] = (__bf16)v.w;
    }
  }
  wait_async0();
  __syncthreads();
  if (!active) return;

  v8f acc[8];
  for (int i = 0; i < 8; ++i) acc[i] = (v8f){0,0,0,0,0,0,0,0};
  int m = lane & 15, kb = (lane >> 4) * 8;
  for (int kt = 0; kt < 8; ++kt) {
    FragU a;
    a.q[0] = *(const uint4*)&stage[w][m][kt * 32 + kb];
    a.q[1] = *(const uint4*)&stage[w][m][kt * 32 + 16 + kb];
    for (int nt = 0; nt < 8; ++nt) {
      FragU b;
      const uint4* bp = (const uint4*)&wlds[(size_t)((kt * 8 + nt) * 32 + lane) * 16];
      b.q[0] = bp[0]; b.q[1] = bp[1];
      acc[nt] = __builtin_amdgcn_wmma_f32_16x16x32_bf16(
          false, a.v, false, b.v, (short)0, acc[nt], false, false);
    }
  }
  int n0 = tile * 16;
  for (int nt = 0; nt < 8; ++nt) {
    int n = nt * 16 + (lane & 15);
    float bias = bo[n];
    for (int r = 0; r < 8; ++r) {
      int M = r + (lane >> 4) * 8;
      float v = acc[nt][r] + bias;
      out[(size_t)(n0 + M) * GNN_HID + n] = v > 0.f ? v : 0.f;
    }
  }
}

// ---------------------------------------------------------------------------
extern "C" void kernel_launch(void* const* d_in, const int* in_sizes, int n_in,
                              void* d_out, int out_size, void* d_ws, size_t ws_size,
                              hipStream_t stream) {
  const float* x     = (const float*)d_in[0];
  const float* eattr = (const float*)d_in[1];
  const float* Wi    = (const float*)d_in[2];
  const float* bi    = (const float*)d_in[3];
  const float* Wh    = (const float*)d_in[4];
  const float* bh    = (const float*)d_in[5];
  const float* Wo    = (const float*)d_in[6];
  const float* bo    = (const float*)d_in[7];
  const int*   ei    = (const int*)d_in[8];
  const int*   rev   = (const int*)d_in[9];
  const int* src = ei;
  const int* dst = ei + GNN_N_EDGES;

  auto al = [](size_t v) { return (v + 255) & ~(size_t)255; };
  char* ws = (char*)d_ws;
  size_t off = 0;
  __bf16* wiF = (__bf16*)(ws + off); off = al(off + (size_t)40 * 32 * 16 * 2);
  __bf16* whF = (__bf16*)(ws + off); off = al(off + (size_t)32 * 32 * 16 * 2);
  __bf16* woF = (__bf16*)(ws + off); off = al(off + (size_t)64 * 32 * 16 * 2);
  __bf16* H0  = (__bf16*)(ws + off); off = al(off + (size_t)GNN_N_EDGES * GNN_HID * 2);
  __bf16* Ha  = (__bf16*)(ws + off); off = al(off + (size_t)GNN_N_EDGES * GNN_HID * 2);
  __bf16* Hb  = (__bf16*)(ws + off); off = al(off + (size_t)GNN_N_EDGES * GNN_HID * 2);
  float*  agg = (float*) (ws + off); off = al(off + (size_t)GNN_N_NODES * GNN_HID * 4);

  const int etiles = GNN_N_EDGES / 16;   // 37500
  const int ntiles = GNN_N_NODES / 16;   // 3125
  const int eblocks = (etiles + 7) / 8;  // 4688
  const int aggquads = GNN_N_NODES * GNN_HID / 4;

  k_pack_weights<<<17, 256, 0, stream>>>(Wi, Wh, Wo, wiF, whF, woF);
  k_edge_h0<<<eblocks, 256, 0, stream>>>(x, eattr, src, bi, wiF, H0, Ha, etiles);

  __bf16* Hc = Ha;
  __bf16* Hn = Hb;
  for (int it = 0; it < 2; ++it) {
    k_zero<<<(aggquads + 255) / 256, 256, 0, stream>>>(agg, aggquads);
    k_scatter<<<(GNN_N_EDGES * 32) / 256, 256, 0, stream>>>(Hc, dst, agg, GNN_N_EDGES * 32);
    k_edge_update<<<eblocks, 256, 0, stream>>>(agg, Hc, H0, src, rev, bh, whF, Hn, etiles);
    __bf16* t = Hc; Hc = Hn; Hn = t;
  }

  k_zero<<<(aggquads + 255) / 256, 256, 0, stream>>>(agg, aggquads);
  k_scatter<<<(GNN_N_EDGES * 32) / 256, 256, 0, stream>>>(Hc, dst, agg, GNN_N_EDGES * 32);
  k_fixup<<<(GNN_N_NODES + 255) / 256, 256, 0, stream>>>(x, agg, GNN_N_NODES);
  k_node_out<<<(ntiles + 1) / 2, 64, 0, stream>>>(x, agg, bo, woF, (float*)d_out, ntiles);
}
